// LocalDecoder_attention_sub_19396072308875
// MI455X (gfx1250) — matrix operand order repaired and checked by reading, not compiled, not run on hardware
//
#include <hip/hip_runtime.h>

typedef unsigned short u16;
typedef __attribute__((ext_vector_type(16))) u16    v16u;
typedef __attribute__((ext_vector_type(16))) __bf16 v16bf;
typedef __attribute__((ext_vector_type(8)))  float  v8f;

#define HEADS 4
#define HH 256
#define CC 32
#define RR 64
#define R3 (RR*RR*RR)
#define NPTS 16384     /* b*n = 2*8192 */
#define P1 4           /* points per stage-1 workgroup */
#define ROWS1 112      /* 4*27 = 108 valid rows, padded to 7 row-tiles */
#define ROWS2 64       /* rows per stage-2 workgroup */

__device__ __forceinline__ u16 f2bf(float f) {          // round-half-up: 1 add + hi16
  unsigned u = __builtin_bit_cast(unsigned, f) + 0x8000u;
  return (u16)(u >> 16);
}
__device__ __forceinline__ float bf2f(u16 h) {
  unsigned u = ((unsigned)h) << 16;
  return __builtin_bit_cast(float, u);
}
__device__ __forceinline__ int clampi(int v, int lo, int hi) {
  return v < lo ? lo : (v > hi ? hi : v);
}
// A-fragment K-order per lane is {0..7,16..23} | {8..15,24..31}: swap bits 3<->4.
__device__ __forceinline__ int kperm(int k) {
  return (k & ~24) | ((k & 8) << 1) | ((k & 16) >> 1);
}
__device__ __forceinline__ v8f wmma_bf16(v16u a, v16u b, v8f c) {
  return __builtin_amdgcn_wmma_f32_16x16x32_bf16(
      false, __builtin_bit_cast(v16bf, a),
      false, __builtin_bit_cast(v16bf, b),
      (short)0, c, false, false);
}

// ------------------------------------------------------------------
// Weight convert + shuffle into per-lane-contiguous fragment order.
// dst[(ct*(K/32)+kt)*512 + lane*16 + e] = bf16(src[(kt*32 + kperm(e + (lane>>4)*16))*256
//                                                  + ct*16 + (lane&15)])
// ------------------------------------------------------------------
__global__ void cvtshuf_kernel(const float* __restrict__ src, u16* __restrict__ dst,
                               int K, int nmat) {
  int i = blockIdx.x * 256 + threadIdx.x;
  int per = K << 8;
  if (i >= per * nmat) return;
  int mat = i / per;
  int j = i - mat * per;
  int tile = j >> 9, r = j & 511;
  int lane = r >> 4, e = r & 15;
  int ktiles = K >> 5;
  int ct = tile / ktiles, kt = tile - ct * ktiles;
  int n  = (ct << 4) + (lane & 15);
  int kk = (kt << 5) + kperm(e + ((lane >> 4) << 4));
  dst[i] = f2bf(src[mat * per + kk * 256 + n]);
}

// ------------------------------------------------------------------ stage 1
// One workgroup (8 waves) handles P1=4 query points. Rows = p*27+nb (108 valid).
__global__ __launch_bounds__(256) void stage1_kernel(
    const float* __restrict__ p, const float* __restrict__ c_grid,
    const float* __restrict__ Wq, const float* __restrict__ bq,
    const float* __restrict__ Wdi0, const float* __restrict__ bdi0,
    const float* __restrict__ bdi1, const float* __restrict__ bk,
    const float* __restrict__ bv,
    const u16* __restrict__ wdi1s,  // shuffled [256x256]
    const u16* __restrict__ wks,    // shuffled [32x256]
    const u16* __restrict__ wvs,    // shuffled [32x256]
    u16* __restrict__ catt)         // [16384][256] bf16 (linear)
{
  __shared__ __align__(32) u16 nf_s[ROWS1][CC];   // 7 KB   (kperm'd cols)
  __shared__ __align__(32) u16 h_s[ROWS1][HH];    // 56 KB  (kperm'd cols)
  __shared__ u16   k_s[ROWS1][HH];                // 56 KB  (linear)
  __shared__ u16   v_s[ROWS1][HH];                // 56 KB  (linear)
  __shared__ float q_s[P1][HH];                   // 4 KB
  __shared__ float interp_s[P1][CC];
  __shared__ float att_s[P1][HEADS][27];
  __shared__ float co_s[P1][3];
  __shared__ int   id_s[P1][3];

  const int pt0 = blockIdx.x * P1;
  const int t = threadIdx.x;

  // ---- per-point coordinates ----
  if (t < P1 * 3) {
    int pp = t / 3, d = t - pp * 3;
    const float invs = 1.0f / (1.0f + 0.1f + 1e-3f);
    float v = p[(pt0 + pp) * 3 + d] * invs + 0.5f;
    v = fminf(fmaxf(v, 0.0f), 1.0f - 1e-3f);
    float c = v * (float)(RR - 1);
    co_s[pp][d] = c;
    id_s[pp][d] = (int)rintf(c);
  }
  __syncthreads();

  // ---- gather 27 neighbor features per point (bf16, kperm'd cols) ----
  for (int e = t; e < ROWS1 * CC; e += 256) {
    int row = e >> 5, c = e & 31;
    float val = 0.0f;
    if (row < P1 * 27) {
      int pp = row / 27, nb = row - pp * 27;
      int ox = nb % 3 - 1, oy = (nb / 3) % 3 - 1, oz = nb / 9 - 1;
      int nx = clampi(id_s[pp][0] + ox, 0, RR - 1);
      int ny = clampi(id_s[pp][1] + oy, 0, RR - 1);
      int nz = clampi(id_s[pp][2] + oz, 0, RR - 1);
      int batch = (pt0 + pp) >> 13;
      val = c_grid[(size_t)(batch * CC + c) * R3 + (nz + RR * ny + RR * RR * nx)];
    }
    nf_s[row][kperm(c)] = f2bf(val);
  }

  // ---- h = relu(d_i @ Wdi0 + bdi0) (kperm'd cols); coalesced Wdi0 reads ----
  for (int e = t; e < ROWS1 * HH; e += 256) {
    int row = e >> 8, k = e & 255;
    float hv = 0.0f;
    if (row < P1 * 27) {
      int pp = row / 27, nb = row - pp * 27;
      int ox = nb % 3 - 1, oy = (nb / 3) % 3 - 1, oz = nb / 9 - 1;
      float dx = co_s[pp][0] - (float)clampi(id_s[pp][0] + ox, 0, RR - 1);
      float dy = co_s[pp][1] - (float)clampi(id_s[pp][1] + oy, 0, RR - 1);
      float dz = co_s[pp][2] - (float)clampi(id_s[pp][2] + oz, 0, RR - 1);
      hv = dx * Wdi0[k] + dy * Wdi0[HH + k] + dz * Wdi0[2 * HH + k] + bdi0[k];
      hv = fmaxf(hv, 0.0f);
    }
    h_s[row][kperm(k)] = f2bf(hv);
  }

  // ---- trilinear interpolation (one thread per (point, channel)) ----
  if (t < P1 * CC) {
    int pp = t >> 5, c = t & 31;
    float cx = co_s[pp][0], cy = co_s[pp][1], cz = co_s[pp][2];
    int x0 = clampi((int)floorf(cx), 0, RR - 1), x1 = clampi(x0 + 1, 0, RR - 1);
    int y0 = clampi((int)floorf(cy), 0, RR - 1), y1 = clampi(y0 + 1, 0, RR - 1);
    int z0 = clampi((int)floorf(cz), 0, RR - 1), z1 = clampi(z0 + 1, 0, RR - 1);
    float wx = cx - floorf(cx), wy = cy - floorf(cy), wz = cz - floorf(cz);
    int batch = (pt0 + pp) >> 13;
    const float* ch = c_grid + (size_t)(batch * CC + c) * R3;
    auto gv = [&](int a, int b, int c2) { return ch[a + RR * b + RR * RR * c2]; };
    interp_s[pp][c] =
        gv(x0, y0, z0) * (1 - wx) * (1 - wy) * (1 - wz) +
        gv(x1, y0, z0) * wx * (1 - wy) * (1 - wz) +
        gv(x0, y1, z0) * (1 - wx) * wy * (1 - wz) +
        gv(x1, y1, z0) * wx * wy * (1 - wz) +
        gv(x0, y0, z1) * (1 - wx) * (1 - wy) * wz +
        gv(x1, y0, z1) * wx * (1 - wy) * wz +
        gv(x0, y1, z1) * (1 - wx) * wy * wz +
        gv(x1, y1, z1) * wx * wy * wz;
  }
  __syncthreads();

  // ---- q = interp @ Wq + bq (Wq loads shared across the 4 points) ----
  {
    float s0 = bq[t], s1 = s0, s2 = s0, s3 = s0;
    #pragma unroll 4
    for (int c = 0; c < CC; ++c) {
      float w = Wq[c * HH + t];
      s0 += interp_s[0][c] * w;
      s1 += interp_s[1][c] * w;
      s2 += interp_s[2][c] * w;
      s3 += interp_s[3][c] * w;
    }
    q_s[0][t] = s0; q_s[1][t] = s1; q_s[2][t] = s2; q_s[3][t] = s3;
  }
  __syncthreads();

  // ---- WMMA: B fragments cached in VGPRs, reused over 7 row-tiles ----
  const int wave = t >> 5, lane = t & 31;
  const int half = lane >> 4, nl = lane & 15;
  for (int ct = wave; ct < 16; ct += 8) {
    const int n = ct * 16 + nl;
    v16u bfr[8];
    #pragma unroll
    for (int kt = 0; kt < 8; ++kt)
      bfr[kt] = *(const v16u*)&wdi1s[((ct << 3) + kt) * 512 + (lane << 4)];
    v16u bkf = *(const v16u*)&wks[ct * 512 + (lane << 4)];
    v16u bvf = *(const v16u*)&wvs[ct * 512 + (lane << 4)];
    const float bd = bdi1[n], bkn = bk[n], bvn = bv[n];
    for (int rt = 0; rt < ROWS1 / 16; ++rt) {
      const int m = rt * 16 + nl;
      v8f acc;
      #pragma unroll
      for (int r = 0; r < 8; ++r) acc[r] = bd;
      #pragma unroll
      for (int kt = 0; kt < 8; ++kt) {
        v16u a = *(const v16u*)&h_s[m][kt * 32 + half * 16];
        acc = wmma_bf16(a, bfr[kt], acc);
      }
      v16u anf = *(const v16u*)&nf_s[m][half * 16];
      v8f acck, accv;
      #pragma unroll
      for (int r = 0; r < 8; ++r) { acck[r] = acc[r] + bkn; accv[r] = acc[r] + bvn; }
      acck = wmma_bf16(anf, bkf, acck);
      accv = wmma_bf16(anf, bvf, accv);
      #pragma unroll
      for (int r = 0; r < 8; ++r) {
        k_s[rt * 16 + r + half * 8][n] = f2bf(acck[r]);
        v_s[rt * 16 + r + half * 8][n] = f2bf(accv[r]);
      }
    }
  }
  __syncthreads();

  // ---- attention logits: (point, head, neighbor) ----
  for (int idx = t; idx < P1 * HEADS * 27; idx += 256) {
    int pp = idx / 108, rem = idx - pp * 108;
    int head = rem / 27, nb = rem - head * 27;
    int row = pp * 27 + nb;
    float s = 0.0f;
    #pragma unroll
    for (int blk = 0; blk < 4; ++blk) {
      v16u kv = *(const v16u*)&k_s[row][head * 64 + blk * 16];
      #pragma unroll
      for (int e = 0; e < 16; ++e)
        s += q_s[pp][head * 64 + blk * 16 + e] * bf2f(kv[e]);
    }
    att_s[pp][head][nb] = s * 0.125f;   // 1/sqrt(64)
  }
  __syncthreads();
  if (t < P1 * HEADS) {
    int pp = t >> 2, head = t & 3;
    float mx = -1e30f;
    for (int nb = 0; nb < 27; ++nb) mx = fmaxf(mx, att_s[pp][head][nb]);
    float sum = 0.0f;
    for (int nb = 0; nb < 27; ++nb) {
      float e = __expf(att_s[pp][head][nb] - mx);
      att_s[pp][head][nb] = e;
      sum += e;
    }
    float inv = 1.0f / sum;
    for (int nb = 0; nb < 27; ++nb) att_s[pp][head][nb] *= inv;
  }
  __syncthreads();
  // ---- c_att = att @ v ----
  {
    int head = t >> 6;
    #pragma unroll
    for (int pp = 0; pp < P1; ++pp) {
      float s = 0.0f;
      #pragma unroll 9
      for (int nb = 0; nb < 27; ++nb)
        s += att_s[pp][head][nb] * bf2f(v_s[pp * 27 + nb][t]);
      catt[(size_t)(pt0 + pp) * HH + t] = f2bf(s);
    }
  }
}

// ------------------------------------------------------------------ stage 2
// A: LDS [ROWS2][256] kperm'd. bfr: 8 cached B fragments for this col-tile.
__device__ __forceinline__ v8f gemm_rt(const u16* A, const v16u* bfr,
                                       float binit, int rt, int nl, int half) {
  v8f acc;
  #pragma unroll
  for (int r = 0; r < 8; ++r) acc[r] = binit;
  #pragma unroll
  for (int kt = 0; kt < 8; ++kt) {
    v16u a = *(const v16u*)&A[(rt * 16 + nl) * 256 + kt * 32 + half * 16];
    acc = wmma_bf16(a, bfr[kt], acc);
  }
  return acc;
}
__device__ __forceinline__ void loadB(v16u* bfr, const u16* __restrict__ Bs,
                                      int ct, int lane) {
  #pragma unroll
  for (int kt = 0; kt < 8; ++kt)
    bfr[kt] = *(const v16u*)&Bs[((ct << 3) + kt) * 512 + (lane << 4)];
}

__global__ __launch_bounds__(256) void stage2_kernel(
    const u16* __restrict__ cattg,
    const u16* __restrict__ wint, const u16* __restrict__ wc,
    const u16* __restrict__ wb0, const u16* __restrict__ wb1,
    const float* __restrict__ bint, const float* __restrict__ bc,
    const float* __restrict__ bb0, const float* __restrict__ bb1,
    const float* __restrict__ Wout, const float* __restrict__ bout,
    float* __restrict__ out)
{
  __shared__ __align__(32) u16 cA[ROWS2 * 256];   // 32 KB (kperm'd)
  __shared__ float net[ROWS2 * 256];              // 64 KB (linear)
  __shared__ __align__(32) u16 aA[ROWS2 * 256];   // 32 KB (kperm'd)
  __shared__ __align__(32) u16 aB[ROWS2 * 256];   // 32 KB (kperm'd)
  const int row0 = blockIdx.x * ROWS2;
  const int t = threadIdx.x;
  const int wave = t >> 5, lane = t & 31;
  const int half = lane >> 4, nl = lane & 15;

  for (int e = t; e < ROWS2 * 256; e += 256) {
    int row = e >> 8, col = e & 255;
    cA[row * 256 + kperm(col)] = cattg[(size_t)row0 * 256 + e];
  }
  __syncthreads();

  // net = c_att @ Wint + bint
  for (int ct = wave; ct < 16; ct += 8) {
    int n = ct * 16 + nl;
    v16u bfr[8];
    loadB(bfr, wint, ct, lane);
    for (int rt = 0; rt < ROWS2 / 16; ++rt) {
      v8f acc = gemm_rt(cA, bfr, bint[n], rt, nl, half);
      #pragma unroll
      for (int r = 0; r < 8; ++r) net[(rt * 16 + r + half * 8) * 256 + n] = acc[r];
    }
  }
  __syncthreads();

  for (int i = 0; i < 5; ++i) {
    // net += c_att @ Wc[i] + bc[i]
    for (int ct = wave; ct < 16; ct += 8) {
      int n = ct * 16 + nl;
      v16u bfr[8];
      loadB(bfr, wc + i * 65536, ct, lane);
      for (int rt = 0; rt < ROWS2 / 16; ++rt) {
        v8f acc = gemm_rt(cA, bfr, bc[i * 256 + n], rt, nl, half);
        #pragma unroll
        for (int r = 0; r < 8; ++r) net[(rt * 16 + r + half * 8) * 256 + n] += acc[r];
      }
    }
    __syncthreads();
    // aA = bf16(relu(net)), kperm'd
    for (int e = t; e < ROWS2 * 256; e += 256) {
      int row = e >> 8, col = e & 255;
      aA[row * 256 + kperm(col)] = f2bf(fmaxf(net[e], 0.0f));
    }
    __syncthreads();
    // aB = bf16(relu(aA @ Wb0[i] + bb0[i])), kperm'd columns
    for (int ct = wave; ct < 16; ct += 8) {
      int n = ct * 16 + nl;
      int pn = kperm(n);
      v16u bfr[8];
      loadB(bfr, wb0 + i * 65536, ct, lane);
      for (int rt = 0; rt < ROWS2 / 16; ++rt) {
        v8f acc = gemm_rt(aA, bfr, bb0[i * 256 + n], rt, nl, half);
        #pragma unroll
        for (int r = 0; r < 8; ++r)
          aB[(rt * 16 + r + half * 8) * 256 + pn] = f2bf(fmaxf(acc[r], 0.0f));
      }
    }
    __syncthreads();
    // net += aB @ Wb1[i] + bb1[i]
    for (int ct = wave; ct < 16; ct += 8) {
      int n = ct * 16 + nl;
      v16u bfr[8];
      loadB(bfr, wb1 + i * 65536, ct, lane);
      for (int rt = 0; rt < ROWS2 / 16; ++rt) {
        v8f acc = gemm_rt(aB, bfr, bb1[i * 256 + n], rt, nl, half);
        #pragma unroll
        for (int r = 0; r < 8; ++r) net[(rt * 16 + r + half * 8) * 256 + n] += acc[r];
      }
    }
    __syncthreads();
  }

  // out = relu(net) @ Wout + bout   (4 lanes per row, shuffle reduce width 4)
  {
    int row = t >> 2, j = t & 3;
    float s = 0.0f;
    #pragma unroll 8
    for (int c = j * 64; c < j * 64 + 64; ++c)
      s += fmaxf(net[row * 256 + c], 0.0f) * Wout[c];
    s += __shfl_xor(s, 1, 4);
    s += __shfl_xor(s, 2, 4);
    if (j == 0) out[row0 + row] = s + bout[0];
  }
}

// ------------------------------------------------------------------ launcher
extern "C" void kernel_launch(void* const* d_in, const int* in_sizes, int n_in,
                              void* d_out, int out_size, void* d_ws, size_t ws_size,
                              hipStream_t stream) {
  (void)in_sizes; (void)n_in; (void)out_size; (void)ws_size;
  const float* p    = (const float*)d_in[0];
  const float* cgrd = (const float*)d_in[1];
  const float* Wq   = (const float*)d_in[2];
  const float* bq   = (const float*)d_in[3];
  const float* Wk   = (const float*)d_in[4];
  const float* bk   = (const float*)d_in[5];
  const float* Wv   = (const float*)d_in[6];
  const float* bv   = (const float*)d_in[7];
  const float* Wdi0 = (const float*)d_in[8];
  const float* bdi0 = (const float*)d_in[9];
  const float* Wdi1 = (const float*)d_in[10];
  const float* bdi1 = (const float*)d_in[11];
  const float* Wint = (const float*)d_in[12];
  const float* bint = (const float*)d_in[13];
  const float* Wc   = (const float*)d_in[14];
  const float* bc   = (const float*)d_in[15];
  const float* Wb0  = (const float*)d_in[16];
  const float* bb0  = (const float*)d_in[17];
  const float* Wb1  = (const float*)d_in[18];
  const float* bb1  = (const float*)d_in[19];
  const float* Wout = (const float*)d_in[20];
  const float* bout = (const float*)d_in[21];
  float* out = (float*)d_out;

  // workspace layout (u16 units)
  u16* ws    = (u16*)d_ws;
  u16* wdi1  = ws;                 // 65536
  u16* wk    = wdi1 + 65536;       // 8192
  u16* wv    = wk + 8192;          // 8192
  u16* wint  = wv + 8192;          // 65536
  u16* wcb   = wint + 65536;       // 5*65536
  u16* wb0b  = wcb + 5 * 65536;    // 5*65536
  u16* wb1b  = wb0b + 5 * 65536;   // 5*65536
  u16* catt  = wb1b + 5 * 65536;   // 16384*256

  cvtshuf_kernel<<<(65536 + 255) / 256, 256, 0, stream>>>(Wdi1, wdi1, 256, 1);
  cvtshuf_kernel<<<(8192 + 255) / 256, 256, 0, stream>>>(Wk, wk, 32, 1);
  cvtshuf_kernel<<<(8192 + 255) / 256, 256, 0, stream>>>(Wv, wv, 32, 1);
  cvtshuf_kernel<<<(65536 + 255) / 256, 256, 0, stream>>>(Wint, wint, 256, 1);
  cvtshuf_kernel<<<(5 * 65536 + 255) / 256, 256, 0, stream>>>(Wc, wcb, 256, 5);
  cvtshuf_kernel<<<(5 * 65536 + 255) / 256, 256, 0, stream>>>(Wb0, wb0b, 256, 5);
  cvtshuf_kernel<<<(5 * 65536 + 255) / 256, 256, 0, stream>>>(Wb1, wb1b, 256, 5);

  stage1_kernel<<<NPTS / P1, 256, 0, stream>>>(p, cgrd, Wq, bq, Wdi0, bdi0, bdi1,
                                               bk, bv, wdi1, wk, wv, catt);
  stage2_kernel<<<NPTS / ROWS2, 256, 0, stream>>>(catt, wint, wcb, wb0b, wb1b,
                                                  bint, bc, bb0, bb1, Wout, bout, out);
}
